// StandardMultiHeadAttention_49125835931924
// MI455X (gfx1250) — compile-verified, matrix operand order checked by reading, not run on hardware
//
#include <hip/hip_runtime.h>

// ---------------------------------------------------------------------------
// Multi-head attention for MI455X (gfx1250), wave32 + WMMA bf16 + TDM.
//   D_MODEL=1024, H=16, depth=64, B=2, S=2048.
// f32->bf16 convert, WMMA GEMM (QKV head-split, Q pre-scaled 1/8), flash
// attention (transposed scores; online softmax), WMMA GEMM out-proj (+bias).
// Row-major tiles (A / Q / K) are staged via the Tensor Data Mover
// (tensor_load_to_lds + s_wait_tensorcnt); k-major operands (B / V) are
// transposed during the cooperative copy so every WMMA fragment gather is two
// contiguous ds_load_b128.
// ---------------------------------------------------------------------------

#define D_MODEL  1024
#define NUM_HEADS 16
#define DEPTH     64
#define BATCH      2
#define SEQ     2048
#define NTOK   (BATCH * SEQ)   // 4096

typedef __attribute__((ext_vector_type(16))) __bf16 v16bf;
typedef __attribute__((ext_vector_type(8)))  float  v8f;
typedef __attribute__((ext_vector_type(4)))  unsigned int v4u;
typedef __attribute__((ext_vector_type(8)))  int v8i_;
typedef __attribute__((ext_vector_type(4)))  int v4i_;

union BF16Frag { v16bf v; unsigned short u[16]; };

#if defined(__gfx1250__) && __has_builtin(__builtin_amdgcn_tensor_load_to_lds)
#define HAVE_TDM 1
#else
#define HAVE_TDM 0
#endif

__device__ __forceinline__ unsigned short f32_to_bf16(float f) {
    unsigned int u = __float_as_uint(f);
    u = (u + 0x7FFFu + ((u >> 16) & 1u)) >> 16;   // round to nearest even
    return (unsigned short)u;
}

// k-index of element e of a 16-bit A/B WMMA fragment (ISA 7.12.2).
__device__ __forceinline__ int frag_k(int e, int hf) {
    return ((e >> 3) << 4) + (hf << 3) + (e & 7);
}

#if HAVE_TDM
// Issue one TDM 2D tile load (bf16 elements) global -> LDS.
//   lds_off          : LDS byte offset of destination (tile stored row-major,
//                      rows of tile_w elements, densely packed)
//   gptr             : global address of tile start
//   tile_w, tile_h   : tile dims in elements (w = contiguous dim)
//   stride_elems     : tensor dim0 stride in elements
__device__ __forceinline__ void tdm_load_2d_bf16(unsigned lds_off,
                                                 const unsigned short* gptr,
                                                 unsigned tile_w, unsigned tile_h,
                                                 unsigned stride_elems) {
    unsigned long long ga = (unsigned long long)(size_t)gptr;
    v4u g0;
    g0[0] = 1u;                                            // count=1, user D#
    g0[1] = lds_off;                                       // lds_addr
    g0[2] = (unsigned)(ga & 0xFFFFFFFFu);                  // global_addr[95:64]
    g0[3] = (unsigned)((ga >> 32) & 0x01FFFFFFu)           // global_addr[120:96]
            | (2u << 30);                                  // type=2 (image)
    v8i_ g1;
    g1[0] = (int)(1u << 16);                               // data_size=1 (2B)
    g1[1] = (int)((tile_w & 0xFFFFu) << 16);               // tensor_dim0 lo
    g1[2] = (int)((tile_w >> 16) | ((tile_h & 0xFFFFu) << 16)); // d0 hi | d1 lo
    g1[3] = (int)((tile_h >> 16) | ((tile_w & 0xFFFFu) << 16)); // d1 hi | tile_dim0
    g1[4] = (int)(tile_h & 0xFFFFu);                       // tile_dim1 (tile_dim2=0)
    g1[5] = (int)stride_elems;                             // tensor_dim0_stride lo
    g1[6] = 0;                                             // stride hi | dim1_stride lo
    g1[7] = 0;
    v4i_ gz4 = {};
    v8i_ gz8 = {};
    // 6-arg form (clang-23 / therock HIP headers): groups 2,3 unused (2D),
    // trailing int32x8 descriptor group zero, cpol 0.
    __builtin_amdgcn_tensor_load_to_lds(g0, g1, gz4, gz4, gz8, 0);
}
#endif

// ---------------------------------------------------------------------------
// f32 -> bf16 elementwise convert (4 elems/thread).
// ---------------------------------------------------------------------------
__global__ void cvt_f32_bf16(const float* __restrict__ src,
                             unsigned short* __restrict__ dst, int n) {
    int i = (blockIdx.x * blockDim.x + threadIdx.x) * 4;
    if (i + 3 < n) {
        float4 f = *(const float4*)(src + i);
        dst[i + 0] = f32_to_bf16(f.x);
        dst[i + 1] = f32_to_bf16(f.y);
        dst[i + 2] = f32_to_bf16(f.z);
        dst[i + 3] = f32_to_bf16(f.w);
    }
}

// ---------------------------------------------------------------------------
// bf16 GEMM: C[M,N] = A[M,K] * B[K,N], f32 accumulate via v_wmma.
// Block tile 128x64, K-step 32, 256 threads = 8 waves (4x2), wave tile 32x32.
// A tile staged by TDM (row-major); B tile transposed into LDS so both
// operand gathers are contiguous 128-bit LDS loads.
// mode 0: bf16 out, head-split scatter [B,H,S,depth], value *= scale
// mode 1: bf16 out, row-major
// mode 2: f32 out + bias, row-major (final projection)
// ---------------------------------------------------------------------------
__global__ __launch_bounds__(256)
void gemm_bf16_wmma(const unsigned short* __restrict__ A,
                    const unsigned short* __restrict__ Bm,
                    int M, int N, int K,
                    unsigned short* __restrict__ outBf,
                    float* __restrict__ outF,
                    const float* __restrict__ bias,
                    int mode, float scale) {
    __shared__ unsigned short As[128 * 32];   // 8 KB, [row][k]
    __shared__ unsigned short Bts[64 * 32];   // 4 KB, [n][k] (transposed)

    const int tid  = threadIdx.x;
    const int wave = tid >> 5, lane = tid & 31;
    const int hf   = lane >> 4, l16 = lane & 15;
    const int wm   = (wave >> 1) * 32;        // wave row in block tile
    const int wn   = (wave & 1) * 32;         // wave col in block tile
    const int rowBase = blockIdx.x * 128;
    const int colBase = blockIdx.y * 64;

    v8f acc[2][2] = {};

    for (int kk = 0; kk < K; kk += 32) {
        // ---- stage A tile (128x32, row-major) ----
#if HAVE_TDM
        if (wave == 0)
            tdm_load_2d_bf16((unsigned)(size_t)As,
                             A + (size_t)rowBase * K + kk, 32u, 128u, (unsigned)K);
#else
        {
            const unsigned int* Ag = (const unsigned int*)(A + (size_t)rowBase * K + kk);
            const int ldau = K >> 1;
#pragma unroll
            for (int i = 0; i < 8; ++i) {
                int idx = tid * 8 + i;                 // 2048 dwords total
                int r = idx >> 4, c = idx & 15;
                ((unsigned int*)As)[idx] = Ag[(size_t)r * ldau + c];
            }
        }
#endif
        // ---- stage B tile (32x64) transposed -> Bts[n][k] ----
        {
            const unsigned int* Bg = (const unsigned int*)(Bm + (size_t)kk * N + colBase);
            const int ldbu = N >> 1;
#pragma unroll
            for (int i = 0; i < 4; ++i) {
                int idx = tid * 4 + i;                 // 1024 dwords total
                int r = idx >> 5, cp = idx & 31;       // k row, dword in row
                unsigned w = Bg[(size_t)r * ldbu + cp];
                Bts[(cp * 2 + 0) * 32 + r] = (unsigned short)(w & 0xFFFFu);
                Bts[(cp * 2 + 1) * 32 + r] = (unsigned short)(w >> 16);
            }
        }
#if HAVE_TDM
        if (wave == 0) __builtin_amdgcn_s_wait_tensorcnt(0);
#endif
        __syncthreads();

        BF16Frag a[2], b[2];
#pragma unroll
        for (int t = 0; t < 2; ++t) {
            const int ar = wm + t * 16 + l16;          // A row (M)
            const int bc = wn + t * 16 + l16;          // B col (N)
#pragma unroll
            for (int e = 0; e < 16; ++e) {
                const int k = frag_k(e, hf);
                a[t].u[e] = As[ar * 32 + k];           // contiguous per 8-group
                b[t].u[e] = Bts[bc * 32 + k];          // contiguous per 8-group
            }
        }
#pragma unroll
        for (int tm = 0; tm < 2; ++tm)
#pragma unroll
            for (int tn = 0; tn < 2; ++tn)
                acc[tm][tn] = __builtin_amdgcn_wmma_f32_16x16x32_bf16(
                    false, a[tm].v, false, b[tn].v,
                    (short)0, acc[tm][tn], false, false);
        __syncthreads();
    }

    // Store (C/D layout: m = hf*8 + j, n = lane&15).
#pragma unroll
    for (int tm = 0; tm < 2; ++tm)
#pragma unroll
        for (int tn = 0; tn < 2; ++tn)
#pragma unroll
            for (int j = 0; j < 8; ++j) {
                const int m = rowBase + wm + tm * 16 + hf * 8 + j;
                const int n = colBase + wn + tn * 16 + l16;
                const float v = acc[tm][tn][j] * scale;
                if (mode == 0) {
                    const int h = n >> 6, d = n & 63;      // depth = 64
                    const int bb = m >> 11, s = m & 2047;  // S = 2048
                    outBf[((((size_t)bb * NUM_HEADS + h) * SEQ + s) << 6) + d] =
                        f32_to_bf16(v);
                } else if (mode == 1) {
                    outBf[(size_t)m * N + n] = f32_to_bf16(v);
                } else {
                    outF[(size_t)m * N + n] = v + bias[n];
                }
            }
}

// ---------------------------------------------------------------------------
// Flash attention per (b,h): Q,K,V in [B,H,S,64] bf16 (Q pre-scaled by 1/8).
// Block = 128 queries (8 waves x 16), streaming 64-key tiles through LDS.
// S^T = K * Q^T and O^T = V^T * P^T: per-query reductions need one
// shfl_xor(16), and P^T's C-register layout is bit-identical to the WMMA
// B-operand layout for the P*V product. Q/K tiles staged by TDM; V tiles
// transposed into Vts[d][key] during the cooperative copy.
// ---------------------------------------------------------------------------
__global__ __launch_bounds__(256)
void flash_attn_wmma(const unsigned short* __restrict__ Q,
                     const unsigned short* __restrict__ Kt,
                     const unsigned short* __restrict__ Vt,
                     unsigned short* __restrict__ concat) {
    __shared__ unsigned short Qs[128 * 64];   // 16 KB, [q][d]
    __shared__ unsigned short Ks[64 * 64];    //  8 KB, [key][d]
    __shared__ unsigned short Vts[64 * 64];   //  8 KB, [d][key] (transposed)

    const int tid  = threadIdx.x;
    const int wave = tid >> 5, lane = tid & 31;
    const int hf   = lane >> 4, l16 = lane & 15;
    const int bh   = blockIdx.y;                  // 0..31 = b*16 + h
    const int qbase = blockIdx.x * 128;
    const size_t hoff = (size_t)bh * SEQ * DEPTH;

    // ---- stage the 128x64 Q tile ----
#if HAVE_TDM
    if (wave == 0) {
        tdm_load_2d_bf16((unsigned)(size_t)Qs, Q + hoff + (size_t)qbase * DEPTH,
                         64u, 128u, 64u);
        __builtin_amdgcn_s_wait_tensorcnt(0);
    }
#else
    {
        const unsigned int* Qg = (const unsigned int*)(Q + hoff + (size_t)qbase * DEPTH);
#pragma unroll
        for (int i = 0; i < 16; ++i)
            ((unsigned int*)Qs)[tid * 16 + i] = Qg[tid * 16 + i];
    }
#endif
    __syncthreads();

    // Q^T B-fragments for this wave's 16 queries (contiguous gathers).
    BF16Frag bq[2];
#pragma unroll
    for (int dh = 0; dh < 2; ++dh)
#pragma unroll
        for (int e = 0; e < 16; ++e)
            bq[dh].u[e] = Qs[(wave * 16 + l16) * 64 + dh * 32 + frag_k(e, hf)];

    float m_i = -1e30f, l_i = 0.0f;
    v8f acco[4] = {};   // O^T accumulators over the 4 d-tiles

    for (int kb = 0; kb < SEQ; kb += 64) {
        __syncthreads();   // previous iteration done with Ks/Vts
        // ---- stage K tile (64x64 row-major) via TDM ----
#if HAVE_TDM
        if (wave == 0)
            tdm_load_2d_bf16((unsigned)(size_t)Ks, Kt + hoff + (size_t)kb * DEPTH,
                             64u, 64u, 64u);
#else
        {
            const unsigned int* Kg = (const unsigned int*)(Kt + hoff + (size_t)kb * DEPTH);
#pragma unroll
            for (int i = 0; i < 8; ++i)
                ((unsigned int*)Ks)[tid * 8 + i] = Kg[tid * 8 + i];
        }
#endif
        // ---- stage V tile transposed -> Vts[d][key] ----
        {
            const unsigned int* Vg = (const unsigned int*)(Vt + hoff + (size_t)kb * DEPTH);
#pragma unroll
            for (int i = 0; i < 8; ++i) {
                int idx = tid * 8 + i;                 // 2048 dwords
                int key = idx >> 5, d2 = (idx & 31) * 2;
                unsigned w = Vg[idx];
                Vts[(d2 + 0) * 64 + key] = (unsigned short)(w & 0xFFFFu);
                Vts[(d2 + 1) * 64 + key] = (unsigned short)(w >> 16);
            }
        }
#if HAVE_TDM
        if (wave == 0) __builtin_amdgcn_s_wait_tensorcnt(0);
#endif
        __syncthreads();

        // S^T tiles: m = key (4 tiles of 16), n = query.
        v8f sacc[4] = {};
#pragma unroll
        for (int kt = 0; kt < 4; ++kt)
#pragma unroll
            for (int dh = 0; dh < 2; ++dh) {
                BF16Frag ak;
#pragma unroll
                for (int e = 0; e < 16; ++e)
                    ak.u[e] = Ks[(kt * 16 + l16) * 64 + dh * 32 + frag_k(e, hf)];
                sacc[kt] = __builtin_amdgcn_wmma_f32_16x16x32_bf16(
                    false, ak.v, false, bq[dh].v, (short)0, sacc[kt], false, false);
            }

        // Online softmax. Query q = lane&15; its 64 scores live in this lane
        // and lane^16, so one shfl_xor completes each reduction.
        float tmax = -1e30f;
#pragma unroll
        for (int kt = 0; kt < 4; ++kt)
#pragma unroll
            for (int j = 0; j < 8; ++j)
                tmax = fmaxf(tmax, sacc[kt][j]);
        tmax = fmaxf(tmax, __shfl_xor(tmax, 16, 32));
        const float m_new = fmaxf(m_i, tmax);
        const float alpha = __expf(m_i - m_new);

        float p[4][8];
        float lsum = 0.0f;
#pragma unroll
        for (int kt = 0; kt < 4; ++kt)
#pragma unroll
            for (int j = 0; j < 8; ++j) {
                p[kt][j] = __expf(sacc[kt][j] - m_new);
                lsum += p[kt][j];
            }
        lsum += __shfl_xor(lsum, 16, 32);
        l_i = l_i * alpha + lsum;
        m_i = m_new;

#pragma unroll
        for (int dt = 0; dt < 4; ++dt)
#pragma unroll
            for (int j = 0; j < 8; ++j)
                acco[dt][j] *= alpha;

        // Pack P^T into B fragments: C-layout element j (key = hf*8+j)
        // coincides with B-layout element e (k = hf*8+e) per 16-key block.
        BF16Frag bp[2];
#pragma unroll
        for (int kh = 0; kh < 2; ++kh)
#pragma unroll
            for (int e = 0; e < 16; ++e)
                bp[kh].u[e] = f32_to_bf16(e < 8 ? p[kh * 2][e] : p[kh * 2 + 1][e - 8]);

        // O^T += V^T * P^T  (A = V^T: m = d, k = key; contiguous in Vts).
#pragma unroll
        for (int dt = 0; dt < 4; ++dt)
#pragma unroll
            for (int kh = 0; kh < 2; ++kh) {
                BF16Frag av;
#pragma unroll
                for (int e = 0; e < 16; ++e)
                    av.u[e] = Vts[(dt * 16 + l16) * 64 + kh * 32 + frag_k(e, hf)];
                acco[dt] = __builtin_amdgcn_wmma_f32_16x16x32_bf16(
                    false, av.v, false, bp[kh].v, (short)0, acco[dt], false, false);
            }
    }

    // Normalize and scatter into concat [B,S,D_MODEL] bf16.
    const float invl = 1.0f / l_i;
    const int bb = bh >> 4, h = bh & 15;
    const int s = qbase + wave * 16 + l16;
#pragma unroll
    for (int dt = 0; dt < 4; ++dt)
#pragma unroll
        for (int j = 0; j < 8; ++j) {
            const int d = dt * 16 + hf * 8 + j;
            concat[(size_t)(bb * SEQ + s) * D_MODEL + h * DEPTH + d] =
                f32_to_bf16(acco[dt][j] * invl);
        }
}

// ---------------------------------------------------------------------------
// Host-side launcher. Inputs: x, wq, wk, wv, wo, bo (all f32).
// ---------------------------------------------------------------------------
extern "C" void kernel_launch(void* const* d_in, const int* in_sizes, int n_in,
                              void* d_out, int out_size, void* d_ws, size_t ws_size,
                              hipStream_t stream) {
    const float* x  = (const float*)d_in[0];
    const float* wq = (const float*)d_in[1];
    const float* wk = (const float*)d_in[2];
    const float* wv = (const float*)d_in[3];
    const float* wo = (const float*)d_in[4];
    const float* bo = (const float*)d_in[5];
    float* out = (float*)d_out;

    char* ws = (char*)d_ws;
    const size_t NX = (size_t)NTOK * D_MODEL;       // 4M elems
    const size_t NW = (size_t)D_MODEL * D_MODEL;    // 1M elems
    const size_t NH = (size_t)BATCH * NUM_HEADS * SEQ * DEPTH; // 4M elems

    unsigned short* Xb  = (unsigned short*)(ws);
    unsigned short* Wqb = (unsigned short*)(ws + 2 * NX);
    unsigned short* Wkb = Wqb + NW;
    unsigned short* Wvb = Wkb + NW;
    unsigned short* Wob = Wvb + NW;
    unsigned short* Qb  = Wob + NW;
    unsigned short* Kb  = Qb + NH;
    unsigned short* Vb  = Kb + NH;
    unsigned short* Cc  = Vb + NH;     // concat, NX elems; total 48 MB

    // 1) convert inputs to bf16
    cvt_f32_bf16<<<dim3((unsigned)(NX / 1024)), 256, 0, stream>>>(x, Xb, (int)NX);
    cvt_f32_bf16<<<dim3((unsigned)(NW / 1024)), 256, 0, stream>>>(wq, Wqb, (int)NW);
    cvt_f32_bf16<<<dim3((unsigned)(NW / 1024)), 256, 0, stream>>>(wk, Wkb, (int)NW);
    cvt_f32_bf16<<<dim3((unsigned)(NW / 1024)), 256, 0, stream>>>(wv, Wvb, (int)NW);
    cvt_f32_bf16<<<dim3((unsigned)(NW / 1024)), 256, 0, stream>>>(wo, Wob, (int)NW);

    // 2) QKV projections (head-split store; 1/sqrt(depth)=0.125 folded into Q)
    const dim3 gGemm(NTOK / 128, D_MODEL / 64);
    gemm_bf16_wmma<<<gGemm, 256, 0, stream>>>(Xb, Wqb, NTOK, D_MODEL, D_MODEL,
                                              Qb, nullptr, nullptr, 0, 0.125f);
    gemm_bf16_wmma<<<gGemm, 256, 0, stream>>>(Xb, Wkb, NTOK, D_MODEL, D_MODEL,
                                              Kb, nullptr, nullptr, 0, 1.0f);
    gemm_bf16_wmma<<<gGemm, 256, 0, stream>>>(Xb, Wvb, NTOK, D_MODEL, D_MODEL,
                                              Vb, nullptr, nullptr, 0, 1.0f);

    // 3) flash attention -> concat
    flash_attn_wmma<<<dim3(SEQ / 128, BATCH * NUM_HEADS), 256, 0, stream>>>(
        Qb, Kb, Vb, Cc);

    // 4) output projection + bias -> f32 out
    gemm_bf16_wmma<<<gGemm, 256, 0, stream>>>(Cc, Wob, NTOK, D_MODEL, D_MODEL,
                                              nullptr, out, bo, 2, 1.0f);
}
